// Net_73203422593186
// MI455X (gfx1250) — matrix operand order repaired and checked by reading, not compiled, not run on hardware
//
#include <hip/hip_runtime.h>
#include <hip/hip_bf16.h>

#define BB   16
#define NN   2048
#define KKNB 128
#define PFC  13
#define HID  32
#define PROP 16
#define NT   (BB * NN)        // 32768 nodes total
#define X2S  48               // padded stride for concat(x, f1) rows (45 used)

typedef __attribute__((ext_vector_type(16))) _Float16 v16h;
typedef __attribute__((ext_vector_type(8)))  float    v8f;

// ---------------------------------------------------------------------------
// Stage A: per-node linears  s = x@Ws+bs (f16 out + sum-of-squares), h = x@Wh+bh
// ---------------------------------------------------------------------------
template<int CIN, int XSTRIDE>
__global__ __launch_bounds__(256) void lin_kernel(
    const float* __restrict__ x,
    const float* __restrict__ Ws, const float* __restrict__ bs,
    const float* __restrict__ Wh, const float* __restrict__ bh,
    _Float16* __restrict__ s16, float* __restrict__ sq, float* __restrict__ h)
{
  int i = blockIdx.x * blockDim.x + threadIdx.x;
  if (i >= NT) return;
  float xr[CIN];
  #pragma unroll
  for (int c = 0; c < CIN; ++c) xr[c] = x[(size_t)i * XSTRIDE + c];

  float acc_sq = 0.f;
  #pragma unroll
  for (int j = 0; j < HID; ++j) {
    float a = bs[j];
    #pragma unroll
    for (int c = 0; c < CIN; ++c) a = fmaf(xr[c], Ws[c * HID + j], a);
    _Float16 a16 = (_Float16)a;
    s16[(size_t)i * HID + j] = a16;
    float ar = (float)a16;                 // sq from rounded s -> self-dist == 0
    acc_sq = fmaf(ar, ar, acc_sq);
  }
  sq[i] = acc_sq;

  #pragma unroll
  for (int j = 0; j < PROP; ++j) {
    float a = bh[j];
    #pragma unroll
    for (int c = 0; c < CIN; ++c) a = fmaf(xr[c], Wh[c * PROP + j], a);
    h[(size_t)i * PROP + j] = a;
  }
}

// ---------------------------------------------------------------------------
// Stage B: WMMA distance tile + deterministic 128-smallest select + aggregate
// One workgroup (8 waves) handles 16 rows of one graph. d2 tile in 128KB LDS.
// ---------------------------------------------------------------------------
__global__ __launch_bounds__(256) void grav_kernel(
    const _Float16* __restrict__ s16, const float* __restrict__ sq,
    const float* __restrict__ h, float* __restrict__ agg)
{
  __shared__ float d2s[16 * NN];          // 128 KB (WGP has 320 KB)

  const int lane = threadIdx.x & 31;
  const int wv   = threadIdx.x >> 5;
  const int half = lane >> 4;             // 0: lanes 0-15, 1: lanes 16-31
  const int l15  = lane & 15;
  const int b        = blockIdx.x >> 7;   // graph
  const int rowBase  = (blockIdx.x & 127) << 4;
  const int nodeBase = b * NN;

  // A fragment: 16-bit A 16x32. lane l<16: row M=l, K {0..7,16..23}; +16: K {8..15,24..31}
  v16h afrag;
  {
    const uint4* ps = (const uint4*)(s16 + (size_t)(nodeBase + rowBase + l15) * HID);
    ((uint4*)&afrag)[0] = ps[half];        // halves k0 .. k0+7   (k0 = half*8)
    ((uint4*)&afrag)[1] = ps[half + 2];    // halves k0+16 .. k0+23
  }
  float sqn[8];
  #pragma unroll
  for (int r = 0; r < 8; ++r) sqn[r] = sq[nodeBase + rowBase + r + half * 8];

  // Each wave computes 16 column tiles of 16: full 16 x 2048 d2 tile.
  for (int i = 0; i < 16; ++i) {
    const int colBase = (i * 8 + wv) * 16;
    const int mnode   = nodeBase + colBase + l15;
    v16h bfrag;                            // B 32x16: lanes 0-15 K {0..7,16..23}, lanes 16-31 K {8..15,24..31}
    {
      const uint4* pb = (const uint4*)(s16 + (size_t)mnode * HID);
      ((uint4*)&bfrag)[0] = pb[half];
      ((uint4*)&bfrag)[1] = pb[half + 2];
    }
    const float sqm = sq[mnode];
    v8f c = {0.f, 0.f, 0.f, 0.f, 0.f, 0.f, 0.f, 0.f};
    c = __builtin_amdgcn_wmma_f32_16x16x32_f16(false, afrag, false, bfrag,
                                               (short)0, c, false, false);
    #pragma unroll
    for (int r = 0; r < 8; ++r) {          // C/D: VGPR r -> M = r + half*8, N = l15
      float d2 = fmaxf(sqn[r] + sqm - 2.0f * c[r], 0.0f);
      d2s[(r + half * 8) * NN + colBase + l15] = d2;
    }
  }
  __syncthreads();

  // Each wave selects+aggregates 2 rows. 64 candidates per lane, cached in VGPRs.
  for (int rr = 0; rr < 2; ++rr) {
    const int row = wv * 2 + rr;
    uint v[64];
    #pragma unroll
    for (int j = 0; j < 64; ++j) v[j] = __float_as_uint(d2s[row * NN + lane + j * 32]);

    // Bitwise radix search: largest cur with count(bits < cur) < 128  -> 128th smallest
    uint cur = 0u;
    for (int bit = 30; bit >= 0; --bit) {
      const uint t = cur | (1u << bit);
      int c = 0;
      #pragma unroll
      for (int j = 0; j < 64; ++j) c += (v[j] < t) ? 1 : 0;
      for (int o = 16; o; o >>= 1) c += __shfl_xor(c, o, 32);
      if (c < KKNB) cur = t;
    }

    // Deterministic tie budget via prefix scan (no atomics -> replay-stable)
    int cl = 0, mt = 0;
    #pragma unroll
    for (int j = 0; j < 64; ++j) { cl += (v[j] < cur) ? 1 : 0; mt += (v[j] == cur) ? 1 : 0; }
    int clT = cl;
    for (int o = 16; o; o >>= 1) clT += __shfl_xor(clT, o, 32);
    const int need = KKNB - clT;
    int inc = mt;
    for (int d = 1; d < 32; d <<= 1) {
      int y = __shfl_up(inc, d, 32);
      if (lane >= d) inc += y;
    }
    const int tieBudget = need - (inc - mt);

    float sum[PROP], mx[PROP];
    #pragma unroll
    for (int d = 0; d < PROP; ++d) { sum[d] = 0.f; mx[d] = -3.4e38f; }

    int tcnt = 0;
    #pragma unroll 4
    for (int j = 0; j < 64; ++j) {
      const uint u = v[j];
      bool sel;
      if (u < cur)       sel = true;
      else if (u == cur) { sel = (tcnt < tieBudget); ++tcnt; }
      else               sel = false;
      if (sel) {
        const int   m = lane + j * 32;
        const float w = __expf(-10.f * __uint_as_float(u));
        const float4* hp = (const float4*)(h + (size_t)(nodeBase + m) * PROP);
        float hr[PROP];
        *(float4*)&hr[0]  = hp[0];
        *(float4*)&hr[4]  = hp[1];
        *(float4*)&hr[8]  = hp[2];
        *(float4*)&hr[12] = hp[3];
        #pragma unroll
        for (int d = 0; d < PROP; ++d) {
          const float msg = w * hr[d];
          sum[d] += msg;
          mx[d] = fmaxf(mx[d], msg);
        }
      }
    }

    for (int o = 16; o; o >>= 1) {
      #pragma unroll
      for (int d = 0; d < PROP; ++d) {
        sum[d] += __shfl_xor(sum[d], o, 32);
        mx[d]  = fmaxf(mx[d], __shfl_xor(mx[d], o, 32));
      }
    }
    if (lane == 0) {
      float* out = agg + (size_t)(nodeBase + rowBase + row) * (2 * PROP);
      #pragma unroll
      for (int d = 0; d < PROP; ++d) {
        out[d]        = sum[d] * (1.0f / KKNB);
        out[PROP + d] = mx[d];
      }
    }
  }
}

// ---------------------------------------------------------------------------
// Stage C1: f1 = LN(x@Wo1 + agg@Wo2 + bo2); emit x2 = [x(13) | f1(32)] stride 48
// ---------------------------------------------------------------------------
__global__ __launch_bounds__(256) void post1_kernel(
    const float* __restrict__ x, const float* __restrict__ agg,
    const float* __restrict__ Wo1, const float* __restrict__ Wo2,
    const float* __restrict__ bo2, float* __restrict__ x2)
{
  int i = blockIdx.x * blockDim.x + threadIdx.x;
  if (i >= NT) return;
  float xr[PFC];
  #pragma unroll
  for (int c = 0; c < PFC; ++c) xr[c] = x[(size_t)i * PFC + c];
  float ag[2 * PROP];
  #pragma unroll
  for (int d = 0; d < 2 * PROP; ++d) ag[d] = agg[(size_t)i * (2 * PROP) + d];

  float y[HID];
  #pragma unroll
  for (int j = 0; j < HID; ++j) {
    float a = bo2[j];
    #pragma unroll
    for (int c = 0; c < PFC; ++c) a = fmaf(xr[c], Wo1[c * HID + j], a);
    #pragma unroll
    for (int d = 0; d < 2 * PROP; ++d) a = fmaf(ag[d], Wo2[d * HID + j], a);
    y[j] = a;
  }
  float m = 0.f;
  #pragma unroll
  for (int j = 0; j < HID; ++j) m += y[j];
  m *= (1.0f / HID);
  float vv = 0.f;
  #pragma unroll
  for (int j = 0; j < HID; ++j) { float t = y[j] - m; vv = fmaf(t, t, vv); }
  const float r = rsqrtf(vv * (1.0f / HID) + 1e-5f);

  float* o = x2 + (size_t)i * X2S;
  #pragma unroll
  for (int c = 0; c < PFC; ++c) o[c] = xr[c];
  #pragma unroll
  for (int j = 0; j < HID; ++j) o[PFC + j] = (y[j] - m) * r;
}

// ---------------------------------------------------------------------------
// Stage C2: f2 = LN(x2@Wo1 + agg@Wo2 + bo2); head: 44 -> 16 (elu) -> 4 (elu) -> 1
// ---------------------------------------------------------------------------
__global__ __launch_bounds__(256) void post2_kernel(
    const float* __restrict__ x2, const float* __restrict__ agg,
    const float* __restrict__ Wo1, const float* __restrict__ Wo2,
    const float* __restrict__ bo2,
    const float* __restrict__ W1, const float* __restrict__ b1,
    const float* __restrict__ W2, const float* __restrict__ b2,
    const float* __restrict__ W3, const float* __restrict__ b3,
    float* __restrict__ out)
{
  const int CIN = PFC + HID;   // 45
  int i = blockIdx.x * blockDim.x + threadIdx.x;
  if (i >= NT) return;
  float xr[PFC + HID];
  #pragma unroll
  for (int c = 0; c < CIN; ++c) xr[c] = x2[(size_t)i * X2S + c];
  float ag[2 * PROP];
  #pragma unroll
  for (int d = 0; d < 2 * PROP; ++d) ag[d] = agg[(size_t)i * (2 * PROP) + d];

  float y[HID];
  #pragma unroll
  for (int j = 0; j < HID; ++j) {
    float a = bo2[j];
    #pragma unroll
    for (int c = 0; c < CIN; ++c) a = fmaf(xr[c], Wo1[c * HID + j], a);
    #pragma unroll
    for (int d = 0; d < 2 * PROP; ++d) a = fmaf(ag[d], Wo2[d * HID + j], a);
    y[j] = a;
  }
  float m = 0.f;
  #pragma unroll
  for (int j = 0; j < HID; ++j) m += y[j];
  m *= (1.0f / HID);
  float vv = 0.f;
  #pragma unroll
  for (int j = 0; j < HID; ++j) { float t = y[j] - m; vv = fmaf(t, t, vv); }
  const float r = rsqrtf(vv * (1.0f / HID) + 1e-5f);
  float f2[HID];
  #pragma unroll
  for (int j = 0; j < HID; ++j) f2[j] = (y[j] - m) * r;

  // z = [ x[0..11] | f2[0..31] ]  (44)  -> W1[44,16]
  float t1[16];
  #pragma unroll
  for (int j = 0; j < 16; ++j) {
    float a = b1[j];
    #pragma unroll
    for (int c = 0; c < PFC - 1; ++c) a = fmaf(xr[c], W1[c * 16 + j], a);
    #pragma unroll
    for (int d = 0; d < HID; ++d) a = fmaf(f2[d], W1[(PFC - 1 + d) * 16 + j], a);
    t1[j] = (a > 0.f) ? a : expm1f(a);
  }
  float t2[4];
  #pragma unroll
  for (int j = 0; j < 4; ++j) {
    float a = b2[j];
    #pragma unroll
    for (int c = 0; c < 16; ++c) a = fmaf(t1[c], W2[c * 4 + j], a);
    t2[j] = (a > 0.f) ? a : expm1f(a);
  }
  float z = b3[0];
  #pragma unroll
  for (int c = 0; c < 4; ++c) z = fmaf(t2[c], W3[c], z);
  out[i] = z;
}

// ---------------------------------------------------------------------------
// Host launcher
// ---------------------------------------------------------------------------
extern "C" void kernel_launch(void* const* d_in, const int* in_sizes, int n_in,
                              void* d_out, int out_size, void* d_ws, size_t ws_size,
                              hipStream_t stream)
{
  // Adaptive top-level mapping: x_pfc (425984) and batch_pfc (32768) by size;
  // remaining 20 leaves in JAX pytree order:
  // conv1{Wh,Wo1,Wo2,Ws,bh,bo2,bs}, conv2{same}, head[W1,b1,W2,b2,W3,b3]
  const float* x = nullptr;
  const void* p[20];
  int np = 0;
  for (int i = 0; i < n_in; ++i) {
    if (in_sizes[i] == NT * PFC)      x = (const float*)d_in[i];
    else if (in_sizes[i] == NT)       { /* batch_pfc: equal-size graphs, unused */ }
    else if (np < 20)                 p[np++] = d_in[i];
  }
  const float* Wh1   = (const float*)p[0];   // [13,16]  = 208
  const float* Wo1_1 = (const float*)p[1];   // [13,32]  = 416
  const float* Wo2_1 = (const float*)p[2];   // [32,32]  = 1024
  const float* Ws1   = (const float*)p[3];   // [13,32]  = 416
  const float* bh1   = (const float*)p[4];   // [16]
  const float* bo2_1 = (const float*)p[5];   // [32]
  const float* bs1   = (const float*)p[6];   // [32]
  const float* Wh2   = (const float*)p[7];   // [45,16]  = 720
  const float* Wo1_2 = (const float*)p[8];   // [45,32]  = 1440
  const float* Wo2_2 = (const float*)p[9];   // [32,32]  = 1024
  const float* Ws2   = (const float*)p[10];  // [45,32]  = 1440
  const float* bh2   = (const float*)p[11];  // [16]
  const float* bo2_2 = (const float*)p[12];  // [32]
  const float* bs2   = (const float*)p[13];  // [32]
  const float* W1    = (const float*)p[14];  // [44,16]  = 704
  const float* b1    = (const float*)p[15];  // [16]
  const float* W2    = (const float*)p[16];  // [16,4]   = 64
  const float* b2    = (const float*)p[17];  // [4]
  const float* W3    = (const float*)p[18];  // [4,1]
  const float* b3    = (const float*)p[19];  // [1]

  char* ws = (char*)d_ws;
  _Float16* s16  = (_Float16*)(ws + 0);          //  2 MiB : [NT,32] f16
  float*    sqv  = (float*)(ws + 2097152);        // 128 KiB: [NT]
  float*    hv   = (float*)(ws + 2228224);        //  2 MiB : [NT,16]
  float*    aggv = (float*)(ws + 4325376);        //  4 MiB : [NT,32]
  float*    x2   = (float*)(ws + 8519680);        //  6 MiB : [NT,48]

  const int blocks_node = NT / 256;               // 128
  const int blocks_grav = BB * (NN / 16);         // 2048

  // Layer 1
  lin_kernel<PFC, PFC><<<blocks_node, 256, 0, stream>>>(x, Ws1, bs1, Wh1, bh1, s16, sqv, hv);
  grav_kernel<<<blocks_grav, 256, 0, stream>>>(s16, sqv, hv, aggv);
  post1_kernel<<<blocks_node, 256, 0, stream>>>(x, aggv, Wo1_1, Wo2_1, bo2_1, x2);

  // Layer 2
  lin_kernel<PFC + HID, X2S><<<blocks_node, 256, 0, stream>>>(x2, Ws2, bs2, Wh2, bh2, s16, sqv, hv);
  grav_kernel<<<blocks_grav, 256, 0, stream>>>(s16, sqv, hv, aggv);
  post2_kernel<<<blocks_node, 256, 0, stream>>>(x2, aggv, Wo1_2, Wo2_2, bo2_2,
                                                W1, b1, W2, b2, W3, b3, (float*)d_out);
}
